// GCN_LSTM_88553635708969
// MI455X (gfx1250) — matrix operand order repaired and checked by reading, not compiled, not run on hardware
//
#include <hip/hip_runtime.h>

typedef __attribute__((ext_vector_type(16))) __bf16 v16bf;
typedef __attribute__((ext_vector_type(8)))  float  v8f;
typedef __attribute__((ext_vector_type(4)))  float  f4v;

#define NN   4096
#define HH   32
#define FIN  5
#define ROLLN 200
#define MT   (NN/16)   // 256 row tiles
#define KTN  (NN/32)   // 128 k tiles
#define NW   4         // waves per block (K-split)
#define KPW  (KTN/NW)  // 32 k tiles per wave

// ---------------------------------------------------------------------------
// Pack A (f32 row-major [N,N]) into bf16 WMMA A-fragments.
// Fragment (mt,kt) = 32 lanes x 16 bf16, lane l holds row M = mt*16+(l&15),
// K slots: s<8 -> kt*32 + 8*(l>>4)+s ; s>=8 -> kt*32 + 16 + 8*(l>>4)+(s-8).
// ---------------------------------------------------------------------------
__global__ void pack_A_kernel(const float* __restrict__ A, __bf16* __restrict__ Apack) {
    int tid  = blockIdx.x * 256 + threadIdx.x;     // 1,048,576 total
    int lane = tid & 31;
    int frag = tid >> 5;                           // mt*128 + kt
    int mt = frag >> 7, kt = frag & 127;
    int row = mt * 16 + (lane & 15);
    int kb  = kt * 32 + 8 * (lane >> 4);
    const float* src = A + (size_t)row * NN + kb;
    __bf16* dst = Apack + (size_t)frag * 512 + lane * 16;
#pragma unroll
    for (int s = 0; s < 8; ++s) {
        dst[s]     = (__bf16)src[s];
        dst[8 + s] = (__bf16)src[16 + s];
    }
}

// ---------------------------------------------------------------------------
// Init: S0 = x0 @ Wx (packed as bf16 B-fragments), h=c=0, pack gate weights
// Wcat[k][g] = (k<32 ? W_ih[g][k] : W_hh[g][k-32]) into B-frag layout,
// pack Wcat2[k][j] = (k<32 ? Wh[k][j] : Wc[k-32][j]), bsum = b_ih + b_hh.
// B-frag: lane l -> N = l&15, K = 8*(l>>4) + slot (slot<8), 16 + 8*(l>>4) + slot-8 (slot>=8).
// ---------------------------------------------------------------------------
__global__ void init_misc_kernel(const float* __restrict__ X,
                                 const float* __restrict__ Wx,
                                 const float* __restrict__ W_ih,
                                 const float* __restrict__ W_hh,
                                 const float* __restrict__ Wh,
                                 const float* __restrict__ Wc,
                                 const float* __restrict__ b_ih,
                                 const float* __restrict__ b_hh,
                                 __bf16* __restrict__ Spack0,
                                 float* __restrict__ h, float* __restrict__ c,
                                 __bf16* __restrict__ WcatG,
                                 __bf16* __restrict__ Wcat2,
                                 float* __restrict__ bsum) {
    int b = blockIdx.x, t = threadIdx.x;
    if (b < 512) {                                  // S0 + zero h,c
        int idx = b * 256 + t;                      // n*32 + j
        int n = idx >> 5, j = idx & 31;
        h[idx] = 0.0f; c[idx] = 0.0f;
        const float* xr = X + (size_t)n * FIN;
        float s = 0.0f;
#pragma unroll
        for (int f = 0; f < FIN; ++f) s += xr[f] * Wx[f * HH + j];
        int kt = n >> 5, kk = n & 31;
        int lane = (j & 15) + (((kk >> 3) & 1) << 4);
        int nt = j >> 4, slot = (kk & 7) + ((kk >> 4) << 3);
        Spack0[(((size_t)kt * 2 + nt) * 32 + lane) * 16 + slot] = (__bf16)s;
    } else if (b < 544) {                           // gate-weight B-frags (16 KB)
        int idx  = (b - 512) * 256 + t;             // kt(2) nt(8) lane(32) slot(16)
        int slot = idx & 15, lane = (idx >> 4) & 31;
        int nt   = (idx >> 9) & 7, kt = idx >> 12;
        int K = kt * 32 + (((slot >> 3) << 4) + 8 * (lane >> 4) + (slot & 7));
        int g = nt * 16 + (lane & 15);
        float v = (K < 32) ? W_ih[g * HH + K] : W_hh[g * HH + (K - 32)];
        WcatG[idx] = (__bf16)v;
    } else if (b < 552) {                           // S-recurrence weight B-frags
        int idx = (b - 544) * 256 + t;              // kt(2) nt(2) lane slot
        if (idx < 2048) {
            int slot = idx & 15, lane = (idx >> 4) & 31;
            int nt   = (idx >> 9) & 1, kt = idx >> 10;
            int K = kt * 32 + (((slot >> 3) << 4) + 8 * (lane >> 4) + (slot & 7));
            int j = nt * 16 + (lane & 15);
            float v = (K < 32) ? Wh[K * HH + j] : Wc[(K - 32) * HH + j];
            Wcat2[idx] = (__bf16)v;
        }
    } else {
        if (t < 128) bsum[t] = b_ih[t] + b_hh[t];
    }
}

// ---------------------------------------------------------------------------
// One recurrent step, fully fused. 4 waves per 16-row tile, K split 4 ways.
//   inp = A @ S            (each wave: 32 kt * 2 nt bf16 WMMAs, 4 indep chains)
//   waves 1..3 spill partial accumulators to LDS; wave 0 reduces
//   gates = [inp|h]@WcatG + bsum    (16 WMMAs, wave 0)
//   LSTM pointwise in accumulator registers -> h,c updated in place (wave 0)
//   S_next = x_{t+1}@Wx + [h_new|c_new]@Wcat2 -> packed bf16 frags (waves 0,1)
// ---------------------------------------------------------------------------
__global__ void __launch_bounds__(32 * NW) step_kernel(
        const __bf16* __restrict__ Apack,
        const __bf16* __restrict__ Sin,
        __bf16* __restrict__ Sout,
        float* __restrict__ h, float* __restrict__ c,
        const __bf16* __restrict__ WcatG,
        const __bf16* __restrict__ Wcat2,
        const float* __restrict__ bsum,
        const float* __restrict__ Xnext,
        const float* __restrict__ Wx,
        int has_next) {
    __shared__ float  part_f[(NW - 1) * 16 * 32];  // partial inp from waves 1..3
    __shared__ __bf16 inp_bf[16 * 32];             // reduced inp tile, bf16
    __shared__ __bf16 hcb[16 * 64];                // [h_new | c_new] tile, bf16
    __shared__ float  wx_s[FIN * HH];

    const int lane = threadIdx.x & 31;
    const int wave = threadIdx.x >> 5;
    const int mt = blockIdx.x;
    const int Mbase = mt * 16;
    const int l15 = lane & 15, lh = lane >> 4;

    for (int i = threadIdx.x; i < FIN * HH; i += 32 * NW) wx_s[i] = Wx[i];

    // ---- main GEMM: partial inp[16,32] over this wave's K range ----
    v8f acc0 = {}, acc1 = {}, acc2 = {}, acc3 = {};
    const __bf16* aptr = Apack + ((size_t)mt * KTN + wave * KPW) * 512 + lane * 16;
    const __bf16* sptr = Sin + (size_t)wave * KPW * 1024 + lane * 16;
#pragma unroll 2
    for (int kt = 0; kt < KPW; kt += 2) {
        v16bf a0 = *(const v16bf*)aptr;
        v16bf b0 = *(const v16bf*)sptr;
        v16bf b1 = *(const v16bf*)(sptr + 512);
        v16bf a1 = *(const v16bf*)(aptr + 512);
        v16bf b2 = *(const v16bf*)(sptr + 1024);
        v16bf b3 = *(const v16bf*)(sptr + 1536);
        aptr += 1024; sptr += 2048;
        acc0 = __builtin_amdgcn_wmma_f32_16x16x32_bf16(false, a0, false, b0, (short)0, acc0, false, false);
        acc1 = __builtin_amdgcn_wmma_f32_16x16x32_bf16(false, a0, false, b1, (short)0, acc1, false, false);
        acc2 = __builtin_amdgcn_wmma_f32_16x16x32_bf16(false, a1, false, b2, (short)0, acc2, false, false);
        acc3 = __builtin_amdgcn_wmma_f32_16x16x32_bf16(false, a1, false, b3, (short)0, acc3, false, false);
    }
    acc0 += acc2;
    acc1 += acc3;

    // waves 1..3: spill partials (C-layout element (r,lane) = [M=r+8*lh][l15])
    if (wave != 0) {
        float* p = part_f + (wave - 1) * 512;
#pragma unroll
        for (int r = 0; r < 8; ++r) {
            int M = r + 8 * lh;
            p[M * 32 + l15]      = acc0[r];
            p[M * 32 + 16 + l15] = acc1[r];
        }
    }
    __syncthreads();

    union FU { v16bf v; f4v q[2]; };

    if (wave == 0) {
        // ---- reduce partials into wave 0's accumulators ----
#pragma unroll
        for (int w = 0; w < NW - 1; ++w) {
            const float* p = part_f + w * 512;
#pragma unroll
            for (int r = 0; r < 8; ++r) {
                int M = r + 8 * lh;
                acc0[r] += p[M * 32 + l15];
                acc1[r] += p[M * 32 + 16 + l15];
            }
        }

        // spill reduced inp to LDS (bf16) for A-fragment re-read
#pragma unroll
        for (int r = 0; r < 8; ++r) {
            int M = r + 8 * lh;
            inp_bf[M * 32 + l15]      = (__bf16)acc0[r];
            inp_bf[M * 32 + 16 + l15] = (__bf16)acc1[r];
        }

        // ---- gates = [inp|h] @ Wcat + (b_ih+b_hh) ----
        v8f g[8];
#pragma unroll
        for (int nt = 0; nt < 8; ++nt) {
            float bv = bsum[nt * 16 + l15];
#pragma unroll
            for (int r = 0; r < 8; ++r) g[nt][r] = bv;
        }

        FU fa;   // A-frag of inp (K = 0..31)
        {
            const char* ib = (const char*)inp_bf + l15 * 64 + lh * 16;
            fa.q[0] = *(const f4v*)(ib);
            fa.q[1] = *(const f4v*)(ib + 32);
        }
#pragma unroll
        for (int nt = 0; nt < 8; ++nt)
            g[nt] = __builtin_amdgcn_wmma_f32_16x16x32_bf16(false, fa.v, false,
                        *(const v16bf*)(WcatG + (size_t)nt * 512 + lane * 16),
                        (short)0, g[nt], false, false);

        v16bf hbv;  // A-frag of h_old (K = 32..63), straight from global f32
        {
            const float* hr = h + (size_t)(Mbase + l15) * HH + 8 * lh;
#pragma unroll
            for (int s = 0; s < 8; ++s) {
                hbv[s]     = (__bf16)hr[s];
                hbv[s + 8] = (__bf16)hr[16 + s];
            }
        }
#pragma unroll
        for (int nt = 0; nt < 8; ++nt)
            g[nt] = __builtin_amdgcn_wmma_f32_16x16x32_bf16(false, hbv, false,
                        *(const v16bf*)(WcatG + (size_t)(8 + nt) * 512 + lane * 16),
                        (short)0, g[nt], false, false);

        // ---- LSTM pointwise, all in registers (i/f/g/o share lane+slot) ----
#pragma unroll
        for (int jt = 0; jt < 2; ++jt) {
#pragma unroll
            for (int r = 0; r < 8; ++r) {
                int M = r + 8 * lh;
                int row = Mbase + M;
                int j = jt * 16 + l15;
                float iv = 1.0f / (1.0f + __expf(-g[jt][r]));
                float fv = 1.0f / (1.0f + __expf(-g[2 + jt][r]));
                float gv = tanhf(g[4 + jt][r]);
                float ov = 1.0f / (1.0f + __expf(-g[6 + jt][r]));
                float cold = c[(size_t)row * HH + j];
                float cn = fv * cold + iv * gv;
                float hn = ov * tanhf(cn);
                c[(size_t)row * HH + j] = cn;
                h[(size_t)row * HH + j] = hn;
                hcb[M * 64 + j]      = (__bf16)hn;
                hcb[M * 64 + 32 + j] = (__bf16)cn;
            }
        }
    }
    __syncthreads();

    // ---- next-step S = x@Wx + h_new@Wh + c_new@Wc (waves 0,1: one nt each) --
    if (has_next && wave < 2) {
        const int nt = wave;           // column tile j in [nt*16, nt*16+16)
        v8f s0 = {};
#pragma unroll
        for (int kt = 0; kt < 2; ++kt) {
            FU fh;
            const char* hb = (const char*)hcb + l15 * 128 + kt * 64 + lh * 16;
            fh.q[0] = *(const f4v*)(hb);
            fh.q[1] = *(const f4v*)(hb + 32);
            s0 = __builtin_amdgcn_wmma_f32_16x16x32_bf16(false, fh.v, false,
                    *(const v16bf*)(Wcat2 + (size_t)(kt * 2 + nt) * 512 + lane * 16),
                    (short)0, s0, false, false);
        }
        const int j = nt * 16 + l15;
        float w0 = wx_s[0 * HH + j], w1 = wx_s[1 * HH + j], w2 = wx_s[2 * HH + j];
        float w3 = wx_s[3 * HH + j], w4 = wx_s[4 * HH + j];
#pragma unroll
        for (int r = 0; r < 8; ++r) {
            int M = r + 8 * lh;
            int row = Mbase + M;
            const float* xr = Xnext + (size_t)row * FIN;
            float sv = s0[r] + xr[0] * w0 + xr[1] * w1 + xr[2] * w2 +
                       xr[3] * w3 + xr[4] * w4;
            int ktO   = row >> 5;
            int kk    = row & 31;
            int laneO = (j & 15) + (((kk >> 3) & 1) << 4);
            int slot  = (kk & 7) + ((kk >> 4) << 3);
            Sout[(((size_t)ktO * 2 + nt) * 32 + laneO) * 16 + slot] = (__bf16)sv;
        }
    }
}

// ---------------------------------------------------------------------------
// out[n] = h[n,:] . W_fc[0,:] + b_fc[0]
// ---------------------------------------------------------------------------
__global__ void out_kernel(const float* __restrict__ h,
                           const float* __restrict__ W_fc,
                           const float* __restrict__ b_fc,
                           float* __restrict__ out) {
    int n = blockIdx.x * 256 + threadIdx.x;
    if (n < NN) {
        float s = b_fc[0];
        const float* hr = h + (size_t)n * HH;
#pragma unroll
        for (int j = 0; j < HH; ++j) s += hr[j] * W_fc[j];
        out[n] = s;
    }
}

extern "C" void kernel_launch(void* const* d_in, const int* in_sizes, int n_in,
                              void* d_out, int out_size, void* d_ws, size_t ws_size,
                              hipStream_t stream) {
    const float* X   = (const float*)d_in[0];
    const float* A   = (const float*)d_in[1];
    const float* Wx  = (const float*)d_in[2];
    const float* Wh  = (const float*)d_in[3];
    const float* Wc  = (const float*)d_in[4];
    const float* Wih = (const float*)d_in[5];
    const float* Whh = (const float*)d_in[6];
    const float* bih = (const float*)d_in[7];
    const float* bhh = (const float*)d_in[8];
    const float* Wfc = (const float*)d_in[9];
    const float* bfc = (const float*)d_in[10];

    char* ws = (char*)d_ws;
    size_t off = 0;
    auto alloc = [&](size_t bytes) -> void* {
        off = (off + 255) & ~(size_t)255;
        void* p = ws + off;
        off += bytes;
        return p;
    };
    __bf16* Apack = (__bf16*)alloc((size_t)MT * KTN * 512 * 2);   // 32 MB
    __bf16* Sp0   = (__bf16*)alloc((size_t)KTN * 2 * 512 * 2);    // 256 KB
    __bf16* Sp1   = (__bf16*)alloc((size_t)KTN * 2 * 512 * 2);
    float*  hbuf  = (float*)alloc((size_t)NN * HH * 4);
    float*  cbuf  = (float*)alloc((size_t)NN * HH * 4);
    __bf16* WcatG = (__bf16*)alloc(16 * 512 * 2);
    __bf16* Wcat2 = (__bf16*)alloc(4 * 512 * 2);
    float*  bsum  = (float*)alloc(128 * 4);

    pack_A_kernel<<<4096, 256, 0, stream>>>(A, Apack);
    init_misc_kernel<<<553, 256, 0, stream>>>(X, Wx, Wih, Whh, Wh, Wc, bih, bhh,
                                              Sp0, hbuf, cbuf, WcatG, Wcat2, bsum);
    __bf16* bufs[2] = {Sp0, Sp1};
    for (int t = 0; t < ROLLN; ++t) {
        int hn = (t + 1 < ROLLN) ? 1 : 0;
        const float* Xn = X + (size_t)(hn ? t + 1 : 0) * NN * FIN;
        step_kernel<<<MT, 32 * NW, 0, stream>>>(Apack, bufs[t & 1], bufs[(t + 1) & 1],
                                                hbuf, cbuf, WcatG, Wcat2, bsum, Xn, Wx, hn);
    }
    out_kernel<<<16, 256, 0, stream>>>(hbuf, Wfc, bfc, (float*)d_out);
}
